// QuestionAwareGNN_42399917146527
// MI455X (gfx1250) — matrix operand order repaired and checked by reading, not compiled, not run on hardware
//
#include <hip/hip_runtime.h>
#include <hip/hip_bf16.h>
#include <math.h>

// ---------------------------------------------------------------------------
// Question-aware GNN, fused f16-WMMA pipeline for gfx1250 (MI455X).
// N=100000 nodes, E=640000 edges, D=128, H=256, L=3.
//   edge MLP:  [16,512]x[512,256] -> relu -> [16,256]x[256,128]   (per wave tile)
//   node MLP:  [16,256]x[256,256] -> relu -> [16,256]x[256,128]
// f16 activation storage (halves HBM traffic on 23.3 TB/s), f32 WMMA accum,
// f32 scatter-add atomics. Weights pre-packed into WMMA B-fragment order so
// each lane's B operand is two contiguous b128 loads (L2/L0-resident).
// B fragments are explicitly double-buffered (4 n-tiles per chunk) so the
// loads for chunk i+1 overlap the 4 WMMAs of chunk i instead of the
// load -> s_wait_loadcnt 0 -> wmma serialization the compiler produced before.
// ---------------------------------------------------------------------------

typedef __attribute__((ext_vector_type(16))) _Float16 v16h;
typedef __attribute__((ext_vector_type(8)))  _Float16 v8h;
typedef __attribute__((ext_vector_type(8)))  float    v8f;

#define DD 128   // node / edge feature dim

static __device__ __forceinline__ v8f wmma_f16(v16h a, v16h b, v8f c) {
  // D = A(16x32 f16) * B(32x16 f16) + C(16x16 f32)
  return __builtin_amdgcn_wmma_f32_16x16x32_f16(
      /*neg_a=*/false, a, /*neg_b=*/false, b,
      /*c_mod=*/(short)0, c, /*reuse_a=*/false, /*reuse_b=*/false);
}

static __device__ __forceinline__ v16h cat8(v8h lo, v8h hi) {
  return __builtin_shufflevector(lo, hi, 0,1,2,3,4,5,6,7,8,9,10,11,12,13,14,15);
}

// A fragment (16x32 f16, CDNA5 ISA layout): lane&15 selects row M; the
// upper half-wave holds K offsets +8 inside each 16-wide K group.  Both
// contiguous 8-half runs load as b128.
static __device__ __forceinline__ v16h afrag_f16(const _Float16* rowp, int koff, int hi) {
  const int base = koff + (hi ? 8 : 0);
  v8h lo = *(const v8h*)(rowp + base);
  v8h hh = *(const v8h*)(rowp + base + 16);
  return cat8(lo, hh);
}

// Same A layout but sourced from an f32 row (used for the scatter accumulator).
static __device__ __forceinline__ v16h afrag_f32(const float* rowp, int koff, int hi) {
  const int base = koff + (hi ? 8 : 0);
  v16h a;
#pragma unroll
  for (int i = 0; i < 8; ++i) {
    a[i]     = (_Float16)rowp[base + i];
    a[i + 8] = (_Float16)rowp[base + 16 + i];
  }
  return a;
}

// B fragment from pre-packed weights: 512 halves per (k-step, n-tile) frag,
// 16 contiguous halves per lane -> two b128 loads.
static __device__ __forceinline__ v16h bfrag(const _Float16* wpack, int frag, int lane) {
  const v8h* p = (const v8h*)(wpack + (long)frag * 512 + lane * 16);
  return cat8(p[0], p[1]);
}

// ---------------------------------------------------------------------------
// Edge MLP: per wave, 16 edges. feat = [node_h[src] | node_h[dst] | edge_h | q]
// (K=512, each 32-wide K-step lies inside one 128-wide segment, so A frags
// come straight from the gathered global rows).  GEMM1 -> relu (via LDS
// reshape) -> GEMM2 -> write edge_h (in place) + atomic scatter-add to agg.
// ---------------------------------------------------------------------------
__global__ __launch_bounds__(128) void edge_mlp_kernel(
    const _Float16* __restrict__ nodeh, _Float16* edgeh,
    float* __restrict__ agg, const _Float16* __restrict__ qv,
    const _Float16* __restrict__ w1p, const float* __restrict__ b1,
    const _Float16* __restrict__ w2p, const float* __restrict__ b2,
    const int* __restrict__ src, const int* __restrict__ dst, int nE) {
  __shared__ __align__(16) _Float16 sh_h[4][16 * 256];  // relu hidden, per wave
  __shared__ __align__(16) float    sh_o[4][16 * 128];  // output tile, per wave

  const int wave = threadIdx.x >> 5;
  const int lane = threadIdx.x & 31;
  const int hi   = lane >> 4;
  const int row  = lane & 15;
  const long e0  = ((long)blockIdx.x * 4 + wave) * 16;
  const bool active = e0 < nE;               // wave-uniform: EXEC stays all-1s

  const v8f vz = {0.f, 0.f, 0.f, 0.f, 0.f, 0.f, 0.f, 0.f};

  if (active) {
    const long e = e0 + row;
    const int s = src[e], d = dst[e];
    const _Float16* seg[4] = { nodeh + (long)s * DD, nodeh + (long)d * DD,
                               edgeh + e * DD, qv };
    v8f acc[16];
#pragma unroll
    for (int nt = 0; nt < 16; ++nt) acc[nt] = vz;

    // --- GEMM1: K=512 (16 k-steps), N=256 (16 n-tiles, 4 chunks of 4) ------
    // Software pipeline: B frags double-buffered; A frag prefetched per k-step.
    v16h bb[2][4];
#pragma unroll
    for (int j = 0; j < 4; ++j) bb[0][j] = bfrag(w1p, j, lane);
    v16h a = afrag_f16(seg[0], 0, hi);
    v16h a_next = a;
#pragma unroll
    for (int it = 0; it < 64; ++it) {        // it = ks*4 + chunk
      const int buf = it & 1;
      const int ks  = it >> 2;
      const int ch  = it & 3;
      if (it + 1 < 64) {                     // issue next chunk's B loads first
        const int ks2 = (it + 1) >> 2, ch2 = (it + 1) & 3;
#pragma unroll
        for (int j = 0; j < 4; ++j)
          bb[buf ^ 1][j] = bfrag(w1p, ks2 * 16 + ch2 * 4 + j, lane);
      }
      if (ch == 3 && ks < 15) {              // prefetch next k-step's A frag
        const int ks2 = ks + 1;
        a_next = afrag_f16(seg[ks2 >> 2], (ks2 & 3) * 32, hi);
      }
#pragma unroll
      for (int j = 0; j < 4; ++j)
        acc[ch * 4 + j] = wmma_f16(a, bb[buf][j], acc[ch * 4 + j]);
      if (ch == 3) a = a_next;
    }

    // bias + relu, reshape C-layout -> row-major LDS tile for second GEMM
    _Float16* ht = sh_h[wave];
#pragma unroll
    for (int nt = 0; nt < 16; ++nt) {
      const int n = nt * 16 + row;
      const float bias = b1[n];
#pragma unroll
      for (int r = 0; r < 8; ++r) {
        const float v = acc[nt][r] + bias;
        ht[(r + hi * 8) * 256 + n] = (_Float16)(v > 0.f ? v : 0.f);
      }
    }
  }
  __syncthreads();

  if (active) {
    const _Float16* ht = sh_h[wave];
    v8f a2[8];
#pragma unroll
    for (int nt = 0; nt < 8; ++nt) a2[nt] = vz;

    // --- GEMM2: K=256 (8 k-steps), N=128 (8 n-tiles, 2 chunks of 4) --------
    v16h bb[2][4];
#pragma unroll
    for (int j = 0; j < 4; ++j) bb[0][j] = bfrag(w2p, j, lane);
    v16h a = afrag_f16(ht + row * 256, 0, hi);
    v16h a_next = a;
#pragma unroll
    for (int it = 0; it < 16; ++it) {        // it = ks*2 + chunk
      const int buf = it & 1;
      const int ks  = it >> 1;
      const int ch  = it & 1;
      if (it + 1 < 16) {
        const int ks2 = (it + 1) >> 1, ch2 = (it + 1) & 1;
#pragma unroll
        for (int j = 0; j < 4; ++j)
          bb[buf ^ 1][j] = bfrag(w2p, ks2 * 8 + ch2 * 4 + j, lane);
      }
      if (ch == 1 && ks < 7)
        a_next = afrag_f16(ht + row * 256, (ks + 1) * 32, hi);
#pragma unroll
      for (int j = 0; j < 4; ++j)
        a2[ch * 4 + j] = wmma_f16(a, bb[buf][j], a2[ch * 4 + j]);
      if (ch == 1) a = a_next;
    }

    float* ot = sh_o[wave];
#pragma unroll
    for (int nt = 0; nt < 8; ++nt) {
      const int n = nt * 16 + row;
      const float bias = b2[n];
#pragma unroll
      for (int r = 0; r < 8; ++r)
        ot[(r + hi * 8) * 128 + n] = a2[nt][r] + bias;
    }
  }
  __syncthreads();

  if (active) {
    const float* ot = sh_o[wave];
    for (int m = 0; m < 16; ++m) {
      const long e = e0 + m;
      const int s = src[e], d = dst[e];
      _Float16* eo = edgeh + e * DD;
#pragma unroll
      for (int c = lane; c < DD; c += 32) {
        const float v = ot[m * 128 + c];
        eo[c] = (_Float16)v;                              // edge_h update (f16)
        atomicAdd(&agg[(long)s * DD + c], v);             // scatter to src
        atomicAdd(&agg[(long)d * DD + c], v);             // scatter to dst
      }
    }
  }
}

// ---------------------------------------------------------------------------
// Node MLP: per wave, 16 nodes.  feat = [node_h | agg] (K=256; agg is f32 and
// converted lane-side).  GEMM1 -> relu -> GEMM2 -> node_h_out (f16).
// ---------------------------------------------------------------------------
__global__ __launch_bounds__(128) void node_mlp_kernel(
    const _Float16* __restrict__ nh_in, const float* __restrict__ agg,
    _Float16* __restrict__ nh_out,
    const _Float16* __restrict__ w1p, const float* __restrict__ b1,
    const _Float16* __restrict__ w2p, const float* __restrict__ b2, int nN) {
  __shared__ __align__(16) _Float16 sh_h[4][16 * 256];
  __shared__ __align__(16) _Float16 sh_o[4][16 * 128];

  const int wave = threadIdx.x >> 5;
  const int lane = threadIdx.x & 31;
  const int hi   = lane >> 4;
  const int row  = lane & 15;
  const long n0  = ((long)blockIdx.x * 4 + wave) * 16;
  const bool active = n0 < nN;               // wave-uniform

  const v8f vz = {0.f, 0.f, 0.f, 0.f, 0.f, 0.f, 0.f, 0.f};

  if (active) {
    const long nd = n0 + row;
    const _Float16* hrow = nh_in + nd * DD;
    const float*    arow = agg + nd * DD;
    v8f acc[16];
#pragma unroll
    for (int nt = 0; nt < 16; ++nt) acc[nt] = vz;

    // --- GEMM1: K=256 (8 k-steps: 4 f16 + 4 f32-converted), N=256 ----------
    v16h bb[2][4];
#pragma unroll
    for (int j = 0; j < 4; ++j) bb[0][j] = bfrag(w1p, j, lane);
    v16h a = afrag_f16(hrow, 0, hi);
    v16h a_next = a;
#pragma unroll
    for (int it = 0; it < 32; ++it) {        // it = ks*4 + chunk
      const int buf = it & 1;
      const int ks  = it >> 2;
      const int ch  = it & 3;
      if (it + 1 < 32) {
        const int ks2 = (it + 1) >> 2, ch2 = (it + 1) & 3;
#pragma unroll
        for (int j = 0; j < 4; ++j)
          bb[buf ^ 1][j] = bfrag(w1p, ks2 * 16 + ch2 * 4 + j, lane);
      }
      if (ch == 3 && ks < 7) {
        const int ks2 = ks + 1;
        a_next = (ks2 < 4) ? afrag_f16(hrow, ks2 * 32, hi)
                           : afrag_f32(arow, (ks2 - 4) * 32, hi);
      }
#pragma unroll
      for (int j = 0; j < 4; ++j)
        acc[ch * 4 + j] = wmma_f16(a, bb[buf][j], acc[ch * 4 + j]);
      if (ch == 3) a = a_next;
    }

    _Float16* ht = sh_h[wave];
#pragma unroll
    for (int nt = 0; nt < 16; ++nt) {
      const int n = nt * 16 + row;
      const float bias = b1[n];
#pragma unroll
      for (int r = 0; r < 8; ++r) {
        const float v = acc[nt][r] + bias;
        ht[(r + hi * 8) * 256 + n] = (_Float16)(v > 0.f ? v : 0.f);
      }
    }
  }
  __syncthreads();

  if (active) {
    const _Float16* ht = sh_h[wave];
    v8f a2[8];
#pragma unroll
    for (int nt = 0; nt < 8; ++nt) a2[nt] = vz;

    // --- GEMM2: K=256 (8 k-steps), N=128 (8 n-tiles, 2 chunks of 4) --------
    v16h bb[2][4];
#pragma unroll
    for (int j = 0; j < 4; ++j) bb[0][j] = bfrag(w2p, j, lane);
    v16h a = afrag_f16(ht + row * 256, 0, hi);
    v16h a_next = a;
#pragma unroll
    for (int it = 0; it < 16; ++it) {
      const int buf = it & 1;
      const int ks  = it >> 1;
      const int ch  = it & 1;
      if (it + 1 < 16) {
        const int ks2 = (it + 1) >> 1, ch2 = (it + 1) & 1;
#pragma unroll
        for (int j = 0; j < 4; ++j)
          bb[buf ^ 1][j] = bfrag(w2p, ks2 * 8 + ch2 * 4 + j, lane);
      }
      if (ch == 1 && ks < 7)
        a_next = afrag_f16(ht + row * 256, (ks + 1) * 32, hi);
#pragma unroll
      for (int j = 0; j < 4; ++j)
        a2[ch * 4 + j] = wmma_f16(a, bb[buf][j], a2[ch * 4 + j]);
      if (ch == 1) a = a_next;
    }

    _Float16* ot = sh_o[wave];
#pragma unroll
    for (int nt = 0; nt < 8; ++nt) {
      const int n = nt * 16 + row;
      const float bias = b2[n];
#pragma unroll
      for (int r = 0; r < 8; ++r)
        ot[(r + hi * 8) * 128 + n] = (_Float16)(a2[nt][r] + bias);
    }
  }
  __syncthreads();

  if (active) {
    const _Float16* ot = sh_o[wave];
    for (int m = 0; m < 16; ++m) {
      _Float16* dp = nh_out + (n0 + m) * DD;
#pragma unroll
      for (int c = lane; c < DD; c += 32) dp[c] = ot[m * 128 + c];
    }
  }
}

// ---------------------------------------------------------------------------
// Helper kernels
// ---------------------------------------------------------------------------
__global__ void qproj_kernel(const float* __restrict__ qe, const float* __restrict__ qW,
                             const float* __restrict__ qb, _Float16* __restrict__ qf) {
  const int j = threadIdx.x;              // 128 threads
  float s = qb[j];
  for (int k = 0; k < 768; ++k) s += qe[k] * qW[k * DD + j];
  qf[j] = (_Float16)s;
}

__global__ void f32_to_f16_kernel(const float* __restrict__ in,
                                  _Float16* __restrict__ out, long n) {
  const long i = (long)blockIdx.x * 256 + threadIdx.x;
  if (i < n) out[i] = (_Float16)in[i];
}

__global__ void zero_f32_kernel(float* __restrict__ p, long n) {
  const long i = (long)blockIdx.x * 256 + threadIdx.x;
  if (i < n) p[i] = 0.f;
}

// Pre-pack a row-major [K,Nc] f32 weight into WMMA B-fragment order (f16):
// out[((ks*ntiles+nt)*512) + lane*16 + i] = W[ks*32 + (lane>=16)*8 + (i&7) + (i>>3)*16][nt*16 + (lane&15)]
__global__ void pack_b_kernel(const float* __restrict__ W, _Float16* __restrict__ out,
                              int K, int Nc) {
  const long i = (long)blockIdx.x * 256 + threadIdx.x;
  if (i >= (long)K * Nc) return;
  const int ii     = (int)(i & 15);
  const int lane   = (int)((i >> 4) & 31);
  const int frag   = (int)(i >> 9);
  const int ntiles = Nc >> 4;
  const int ks = frag / ntiles;
  const int nt = frag - ks * ntiles;
  const int k = ks * 32 + (lane >> 4) * 8 + (ii & 7) + (ii >> 3) * 16;
  const int n = nt * 16 + (lane & 15);
  out[i] = (_Float16)W[(long)k * Nc + n];
}

__global__ void score_kernel(const _Float16* __restrict__ h, const float* __restrict__ w,
                             const float* __restrict__ b, float* __restrict__ out, long n) {
  const long i = (long)blockIdx.x * 256 + threadIdx.x;
  if (i >= n) return;
  const _Float16* r = h + i * DD;
  float s = b[0];
#pragma unroll
  for (int k = 0; k < DD; ++k) s += (float)r[k] * w[k];
  out[i] = 1.f / (1.f + __expf(-s));
}

// ---------------------------------------------------------------------------
extern "C" void kernel_launch(void* const* d_in, const int* in_sizes, int n_in,
                              void* d_out, int out_size, void* d_ws, size_t ws_size,
                              hipStream_t stream) {
  const float* node_emb = (const float*)d_in[0];
  const float* rel_emb  = (const float*)d_in[1];
  const float* q_emb    = (const float*)d_in[2];
  const int*   eidx     = (const int*)d_in[3];
  // d_in[4] = edge_type (unused, as in reference)
  const float* qW  = (const float*)d_in[5];
  const float* qb  = (const float*)d_in[6];
  const float* eW1 = (const float*)d_in[7];
  const float* eb1 = (const float*)d_in[8];
  const float* eW2 = (const float*)d_in[9];
  const float* eb2 = (const float*)d_in[10];
  const float* nW1 = (const float*)d_in[11];
  const float* nb1 = (const float*)d_in[12];
  const float* nW2 = (const float*)d_in[13];
  const float* nb2 = (const float*)d_in[14];
  const float* esW = (const float*)d_in[15];
  const float* esb = (const float*)d_in[16];
  const float* nsW = (const float*)d_in[17];
  const float* nsb = (const float*)d_in[18];

  const long N = in_sizes[0] / DD;                 // 100000
  const long E = in_sizes[1] / DD;                 // 640000
  const int  L = in_sizes[7] / (4 * DD * 2 * DD);  // 3

  // ---- workspace carve-up (256B aligned) ----
  char* p = (char*)d_ws;
  auto take = [&](size_t b) -> char* {
    char* r = p;
    p += (b + 255) & ~(size_t)255;
    return r;
  };
  _Float16* nhA = (_Float16*)take((size_t)N * DD * 2);
  _Float16* nhB = (_Float16*)take((size_t)N * DD * 2);
  _Float16* eh  = (_Float16*)take((size_t)E * DD * 2);
  float*    agg = (float*)take((size_t)N * DD * 4);
  _Float16* qf  = (_Float16*)take(512);
  _Float16* ew1p[8]; _Float16* ew2p[8]; _Float16* nw1p[8]; _Float16* nw2p[8];
  for (int l = 0; l < L; ++l) {
    ew1p[l] = (_Float16*)take(512 * 256 * 2);
    ew2p[l] = (_Float16*)take(256 * 128 * 2);
    nw1p[l] = (_Float16*)take(256 * 256 * 2);
    nw2p[l] = (_Float16*)take(256 * 128 * 2);
  }

  const int* src = eidx;
  const int* dst = eidx + E;
  const long nelN = N * DD, nelE = E * DD;

  // ---- one-time prep (per call; everything we read is re-initialized) ----
  qproj_kernel<<<1, 128, 0, stream>>>(q_emb, qW, qb, qf);
  f32_to_f16_kernel<<<(int)((nelN + 255) / 256), 256, 0, stream>>>(node_emb, nhA, nelN);
  f32_to_f16_kernel<<<(int)((nelE + 255) / 256), 256, 0, stream>>>(rel_emb, eh, nelE);
  for (int l = 0; l < L; ++l) {
    pack_b_kernel<<<(512 * 256 + 255) / 256, 256, 0, stream>>>(eW1 + (long)l * 512 * 256, ew1p[l], 512, 256);
    pack_b_kernel<<<(256 * 128 + 255) / 256, 256, 0, stream>>>(eW2 + (long)l * 256 * 128, ew2p[l], 256, 128);
    pack_b_kernel<<<(256 * 256 + 255) / 256, 256, 0, stream>>>(nW1 + (long)l * 256 * 256, nw1p[l], 256, 256);
    pack_b_kernel<<<(256 * 128 + 255) / 256, 256, 0, stream>>>(nW2 + (long)l * 256 * 128, nw2p[l], 256, 128);
  }

  // ---- layers ----
  _Float16* cur = nhA;
  _Float16* nxt = nhB;
  const int eblocks = (int)(((E + 15) / 16 + 3) / 4);   // 4 waves/block, 16 edges/wave
  const int nblocks = (int)(((N + 15) / 16 + 3) / 4);
  for (int l = 0; l < L; ++l) {
    zero_f32_kernel<<<(int)((nelN + 255) / 256), 256, 0, stream>>>(agg, nelN);
    edge_mlp_kernel<<<eblocks, 128, 0, stream>>>(cur, eh, agg, qf,
        ew1p[l], eb1 + (long)l * 256, ew2p[l], eb2 + (long)l * 128, src, dst, (int)E);
    node_mlp_kernel<<<nblocks, 128, 0, stream>>>(cur, agg, nxt,
        nw1p[l], nb1 + (long)l * 256, nw2p[l], nb2 + (long)l * 128, (int)N);
    _Float16* t = cur; cur = nxt; nxt = t;
  }

  // ---- heads: out = [node_scores (N) | edge_scores (E)] ----
  float* out = (float*)d_out;
  score_kernel<<<(int)((N + 255) / 256), 256, 0, stream>>>(cur, nsW, nsb, out, N);
  score_kernel<<<(int)((E + 255) / 256), 256, 0, stream>>>(eh, esW, esb, out + N, E);
}
